// FourierNeuralOperatorNN_51762945851952
// MI455X (gfx1250) — compile-verified, matrix operand order checked by reading, not run on hardware
//
#include <hip/hip_runtime.h>
#include <hip/hip_fp16.h>
#include <stdint.h>

// ---------------------------------------------------------------------------
// FNO-PINN loss on MI455X (gfx1250, wave32, WMMA).
// 5-component derivative jets folded into GEMM rows; spectral conv as
// truncated separable DFT GEMMs; channel contractions via
// V_WMMA_F32_16X16X32_F16; A tiles staged with GLOBAL_LOAD_ASYNC_TO_LDS_B128
// (ASYNCcnt) and double buffering.
// ---------------------------------------------------------------------------

typedef __attribute__((ext_vector_type(16))) _Float16 v16h;
typedef __attribute__((ext_vector_type(8)))  _Float16 v8h;
typedef __attribute__((ext_vector_type(8)))  float    v8f;
typedef int v4i_vec __attribute__((vector_size(4 * sizeof(int))));

constexpr int kS1  = 128;
constexpr int kS2  = 512;
constexpr int NPTS = kS1 * kS2;      // 65536
constexpr int WID  = 50;             // FNO width
constexpr int CP   = 64;             // padded channels
constexpr int NJ   = 5;              // jet: val, dx, dy, dxx, dyy
constexpr int M1C  = 12;
constexpr int M2C  = 12;
constexpr float NU_C = 1.56e-05f;

// ------------------------------ GEMM (WMMA) --------------------------------
// C[M x Ncol] = A[M x K] * B[K x Ncol], A/B f16 row-major, C f32.
// Block = 256 threads = 8 waves; block tile 128x64; K-step 32.
// B staged per 256-wide K chunk (transposed); A tiles double-buffered and
// copied with GLOBAL_LOAD_ASYNC_TO_LDS_B128 when the toolchain exposes it.
constexpr int TM = 128, TN = 64, TK = 32;
constexpr int TKB   = 256;          // B chunk along K
constexpr int LDA_S = 40;           // LDS stride (halves) for A tile (32+8)
constexpr int LDB_S = TKB + 8;      // LDS stride (halves) for B chunk

#if defined(__HIP_DEVICE_COMPILE__) && \
    __has_builtin(__builtin_amdgcn_global_load_async_to_lds_b128) && \
    __has_builtin(__builtin_amdgcn_s_wait_asynccnt)
#define HAS_ASYNC_LDS 1
#else
#define HAS_ASYNC_LDS 0
#endif

__device__ __forceinline__
void stage_A_tile(const _Float16* __restrict__ A, int row0, int gk, int lda,
                  _Float16* dst /*LDS*/, int t)
{
    // TM x TK halves = 512 chunks of 16B; 2 per thread.
#pragma unroll
    for (int i = t; i < TM * TK / 8; i += 256) {
        int r  = i >> 2;
        int cc = (i & 3) * 8;
        const _Float16* gp = &A[(size_t)(row0 + r) * lda + gk + cc];
        _Float16* lp = dst + r * LDA_S + cc;
#if HAS_ASYNC_LDS
        __builtin_amdgcn_global_load_async_to_lds_b128(
            (__attribute__((address_space(1))) v4i_vec*)(uintptr_t)gp,
            (__attribute__((address_space(3))) v4i_vec*)(uintptr_t)lp, 0, 0);
#else
        *(v8h*)lp = *(const v8h*)gp;
#endif
    }
}

__global__ __launch_bounds__(256)
void k_gemm_f16f32(const _Float16* __restrict__ A, const _Float16* __restrict__ B,
                   float* __restrict__ C, int M, int K, int Ncol,
                   int lda, int ldb, int ldc, int accum)
{
    __builtin_amdgcn_s_wait_tensorcnt(0);   // CDNA5 split-counter path

    __shared__ __align__(16) _Float16 As[2][TM * LDA_S];   // 2 x 10240 B
    __shared__ __align__(16) _Float16 Bs[TN * LDB_S];      // 33792 B (transposed)

    const int row0 = blockIdx.x * TM;
    const int col0 = blockIdx.y * TN;
    const int t    = threadIdx.x;
    const int lane = t & 31;
    const int wave = t >> 5;
    const int mrow = lane & 15;   // row-in-strip (A) / column (B,C)
    const int kh   = lane >> 4;   // K half-block select per ISA 16-bit layout

    v8f acc[4];
#pragma unroll
    for (int c16 = 0; c16 < 4; ++c16)
#pragma unroll
        for (int i = 0; i < 8; ++i) acc[c16][i] = 0.f;

    int buf = 0;
    bool primed = false;
    for (int kb = 0; kb < K; kb += TKB) {
        const int kbLen = (K - kb < TKB) ? (K - kb) : TKB;
        if (kb) __syncthreads();     // everyone done reading Bs before restage
        // stage B chunk transposed: Bs[col][k]
        for (int i = t; i < kbLen * TN; i += 256) {
            int kk = i >> 6;         // i / TN
            int cc = i & 63;
            _Float16 vb = (_Float16)0.f;
            if (col0 + cc < Ncol) vb = B[(size_t)(kb + kk) * ldb + col0 + cc];
            Bs[cc * LDB_S + kk] = vb;
        }
        if (kb + TKB < K && t < TN)  // warm next B chunk toward the WGP
            __builtin_prefetch(&B[(size_t)(kb + TKB) * ldb + col0 + t], 0, 1);
        if (!primed) { stage_A_tile(A, row0, kb, lda, As[buf], t); primed = true; }

        for (int k0 = 0; k0 < kbLen; k0 += TK) {
#if HAS_ASYNC_LDS
            __builtin_amdgcn_s_wait_asynccnt(0);   // current tile landed in LDS
#endif
            __syncthreads();   // tile visible to all waves; other buffer reusable

            const int nk = kb + k0 + TK;
            if (nk < K)        // overlap: issue next tile while computing this one
                stage_A_tile(A, row0, nk, lda, As[buf ^ 1], t);

            // A fragment (16x32, 16-bit ISA layout): lane holds row mrow,
            // K = kh*8 + [0..7] and 16 + kh*8 + [0..7]  -> two b128 LDS loads.
            const int aoff = (wave * 16 + mrow) * LDA_S;
            v8h alo = *(const v8h*)(&As[buf][aoff + kh * 8]);
            v8h ahi = *(const v8h*)(&As[buf][aoff + 16 + kh * 8]);
            v16h a;
#pragma unroll
            for (int i = 0; i < 8; ++i) { a[i] = alo[i]; a[8 + i] = ahi[i]; }

#pragma unroll
            for (int c16 = 0; c16 < 4; ++c16) {
                const int boff = (c16 * 16 + mrow) * LDB_S + k0;
                v8h blo = *(const v8h*)(&Bs[boff + kh * 8]);
                v8h bhi = *(const v8h*)(&Bs[boff + 16 + kh * 8]);
                v16h b;
#pragma unroll
                for (int i = 0; i < 8; ++i) { b[i] = blo[i]; b[8 + i] = bhi[i]; }
                acc[c16] = __builtin_amdgcn_wmma_f32_16x16x32_f16(
                    false, a, false, b, (short)0, acc[c16], false, false);
            }
            buf ^= 1;
        }
    }

    // Store: C/D layout — lane => col = lane&15, M = vgpr + 8*(lane>>4).
    const int gr = row0 + wave * 16;
#pragma unroll
    for (int c16 = 0; c16 < 4; ++c16) {
        int col = col0 + c16 * 16 + mrow;
        if (col >= Ncol) continue;
#pragma unroll
        for (int v = 0; v < 8; ++v) {
            int r = gr + v + 8 * kh;
            if (r < M) {
                float val = acc[c16][v];
                if (accum) val += C[(size_t)r * ldc + col];
                C[(size_t)r * ldc + col] = val;
            }
        }
    }
}

// ------------------------------ small kernels ------------------------------

__global__ void k_fill_f16(_Float16* p, int n) {
    int i = blockIdx.x * 256 + threadIdx.x;
    if (i < n) p[i] = (_Float16)0.f;
}

// Pad+convert f32 (R x Ccols, contiguous) -> f16 (Rp x Cp), zero pad.
__global__ void k_pad_f16(const float* __restrict__ src, _Float16* __restrict__ dst,
                          int R, int Ccols, int Rp, int Cp) {
    int i = blockIdx.x * 256 + threadIdx.x;
    if (i >= Rp * Cp) return;
    int r = i / Cp, c = i % Cp;
    dst[i] = (_Float16)((r < R && c < Ccols) ? src[r * Ccols + c] : 0.f);
}

// fc0 lift + jet seed: val = x*w0x + y*w0y + b ; d/dx = w0x ; d/dy = w0y.
__global__ void k_lift(const float* __restrict__ x, const float* __restrict__ y,
                       const float* __restrict__ w0, const float* __restrict__ b0,
                       _Float16* __restrict__ act) {
    int n = blockIdx.x * 256 + threadIdx.x;
    if (n >= NPTS) return;
    float xv = x[n], yv = y[n];
    for (int c = 0; c < CP; ++c) {
        float wx = (c < WID) ? w0[c] : 0.f;
        float wy = (c < WID) ? w0[WID + c] : 0.f;
        float bb = (c < WID) ? b0[c] : 0.f;
        act[((size_t)0 * NPTS + n) * CP + c] = (_Float16)(xv * wx + yv * wy + bb);
        act[((size_t)1 * NPTS + n) * CP + c] = (_Float16)wx;
        act[((size_t)2 * NPTS + n) * CP + c] = (_Float16)wy;
        act[((size_t)3 * NPTS + n) * CP + c] = (_Float16)0.f;
        act[((size_t)4 * NPTS + n) * CP + c] = (_Float16)0.f;
    }
}

// (j,n,c) f16 -> rows (j*CP+c)*S1 + x, cols y (f16), for the y-DFT GEMM.
__global__ void k_perm_fwd(const _Float16* __restrict__ act, _Float16* __restrict__ g1) {
    size_t i = (size_t)blockIdx.x * 256 + threadIdx.x;
    if (i >= (size_t)NJ * CP * kS1 * kS2) return;
    int yy = (int)(i % kS2); size_t r = i / kS2;
    int xg = (int)(r % kS1); size_t jc = r / kS1;
    int c = (int)(jc % CP); int j = (int)(jc / CP);
    int n = xg * kS2 + yy;
    g1[i] = act[((size_t)j * NPTS + n) * CP + c];
}

// Y1 f32 rows (jc*S1+x) x 32  ->  G2 f16 rows (jc*32+q) x 128 (cols = x).
__global__ void k_perm_y1g2(const float* __restrict__ y1, _Float16* __restrict__ g2) {
    size_t i = (size_t)blockIdx.x * 256 + threadIdx.x;
    if (i >= (size_t)NJ * CP * 32 * kS1) return;
    int xg = (int)(i % kS1); size_t r = i / kS1;
    int q = (int)(r % 32); size_t jc = r / 32;
    g2[i] = (_Float16)y1[((size_t)jc * kS1 + xg) * 32 + q];
}

// Y3 f32 rows (jc*32+q) x 256 (col=2x+s) -> G4 f16 rows (jc*S1+x) x 32 (col=q).
__global__ void k_perm_y3g4(const float* __restrict__ y3, _Float16* __restrict__ g4) {
    size_t i = (size_t)blockIdx.x * 256 + threadIdx.x;
    if (i >= (size_t)NJ * CP * kS1 * 32) return;
    int q = (int)(i % 32); size_t r = i / 32;
    int xg = (int)(r % kS1); size_t jc = r / kS1;
    g4[i] = (_Float16)y3[((size_t)jc * 32 + q) * 256 + 2 * xg + (q & 1)];
}

// Truncated-DFT twiddle matrices (f16), built on device each call (cheap).
__global__ void k_twiddles(_Float16* DYF, _Float16* DXF, _Float16* DXI, _Float16* DYI) {
    int i = blockIdx.x * 256 + threadIdx.x;
    const float PI2 = 6.28318530718f;
    if (i < 512 * 32) {               // forward y-DFT: 512 x 32 (24 used)
        int yy = i / 32, col = i % 32, m = col >> 1;
        float v = 0.f;
        if (m < M2C) { float a = PI2 * m * yy / 512.f; v = (col & 1) ? -__sinf(a) : __cosf(a); }
        DYF[i] = (_Float16)v;
    }
    if (i < 128 * 64) {               // forward x-DFT: 128 x 64 (48 used)
        int xg = i / 64, col = i % 64, q = col >> 1;
        float v = 0.f;
        if (q < 24) {
            int kx = (q < 12) ? q : (116 + (q - 12));
            float a = PI2 * kx * xg / 128.f; v = (col & 1) ? -__sinf(a) : __cosf(a);
        }
        DXF[i] = (_Float16)v;
    }
    if (i < 64 * 256) {               // inverse x-DFT: 64 x 256 (rows 48 used)
        int p = i / 256, col = i % 256, q = p >> 1;
        float v = 0.f;
        if (q < 24) {
            int kx = (q < 12) ? q : (116 + (q - 12));
            int xg = col >> 1;
            float a = PI2 * kx * xg / 128.f;
            float cs = __cosf(a) / 128.f, sn = __sinf(a) / 128.f;
            v = (p & 1) ? ((col & 1) ? cs : -sn) : ((col & 1) ? sn : cs);
        }
        DXI[i] = (_Float16)v;
    }
    if (i < 32 * 512) {               // inverse y (irfft, hermitian doubling): 32 x 512
        int q = i / 512, yy = i % 512, m = q >> 1;
        float v = 0.f;
        if (m < M2C) {
            float wm = (m == 0) ? 1.f : 2.f;
            float a = PI2 * m * yy / 512.f;
            v = (q & 1) ? (-wm * __sinf(a) / 512.f) : (wm * __cosf(a) / 512.f);
        }
        DYI[i] = (_Float16)v;
    }
}

// Complex mode mixing: out[o] = sum_i W[i][o][mode] * Z[i], per (jet, mx, my).
__global__ void k_modemix(const float* __restrict__ Y2,
                          const float* __restrict__ w1r, const float* __restrict__ w1i,
                          const float* __restrict__ w2r, const float* __restrict__ w2i,
                          _Float16* __restrict__ G3) {
    int b  = blockIdx.x;
    int my = b % M2C;
    int mx = (b / M2C) % 24;
    int j  = b / (M2C * 24);
    int o  = threadIdx.x;
    __shared__ float zr[WID], zi[WID];
    if (o < WID) {
        int base = (j * CP + o) * 32 + 2 * my;
        float arr = Y2[(size_t)base * 64 + 2 * mx];
        float ari = Y2[(size_t)base * 64 + 2 * mx + 1];
        float air = Y2[(size_t)(base + 1) * 64 + 2 * mx];
        float aii = Y2[(size_t)(base + 1) * 64 + 2 * mx + 1];
        zr[o] = arr - aii;        // combine separable re/im components
        zi[o] = ari + air;
    }
    __syncthreads();
    if (o < WID) {
        const float* wr = (mx < 12) ? w1r : w2r;
        const float* wi = (mx < 12) ? w1i : w2i;
        int mxx = (mx < 12) ? mx : mx - 12;
        float sr = 0.f, si = 0.f;
        for (int ic = 0; ic < WID; ++ic) {
            float wrv = wr[((ic * WID + o) * M1C + mxx) * M2C + my];
            float wiv = wi[((ic * WID + o) * M1C + mxx) * M2C + my];
            sr += zr[ic] * wrv - zi[ic] * wiv;
            si += zr[ic] * wiv + zi[ic] * wrv;
        }
        int ob = (j * CP + o) * 32 + 2 * my;   // re-split into separable components
        G3[(size_t)ob * 64 + 2 * mx]       = (_Float16)sr;
        G3[(size_t)ob * 64 + 2 * mx + 1]   = (_Float16)si;
        G3[(size_t)(ob + 1) * 64 + 2 * mx] = (_Float16)si;
    }
}

// x1 (spectral, (j,c,x,y) f32) + x2 (pointwise, (j,n,c) f32) + bias, jet GELU.
__global__ void k_combine_gelu(const float* __restrict__ X1, const float* __restrict__ X2,
                               const float* __restrict__ bias, _Float16* __restrict__ out,
                               int applyGelu) {
    int idx = blockIdx.x * 256 + threadIdx.x;
    if (idx >= NPTS * CP) return;
    int n = idx / CP, c = idx % CP;
    if (c >= WID) {
        for (int j = 0; j < NJ; ++j) out[((size_t)j * NPTS + n) * CP + c] = (_Float16)0.f;
        return;
    }
    int xg = n / kS2, yy = n % kS2;
    float v[NJ];
    for (int j = 0; j < NJ; ++j) {
        float a = X1[(((size_t)j * CP + c) * kS1 + xg) * kS2 + yy];
        float b = X2[((size_t)j * NPTS + n) * CP + c];
        v[j] = a + b + ((j == 0) ? bias[c] : 0.f);
    }
    if (applyGelu) {
        float z   = v[0];
        float Phi = 0.5f * (1.f + erff(z * 0.70710678f));
        float phi = 0.39894228f * __expf(-0.5f * z * z);
        float g0  = z * Phi;
        float g1  = Phi + z * phi;
        float g2  = 2.f * phi - z * z * phi;
        float dx = v[1], dy = v[2], dxx = v[3], dyy = v[4];
        v[0] = g0;
        v[1] = g1 * dx;
        v[2] = g1 * dy;
        v[3] = g2 * dx * dx + g1 * dxx;
        v[4] = g2 * dy * dy + g1 * dyy;
    }
    for (int j = 0; j < NJ; ++j) out[((size_t)j * NPTS + n) * CP + c] = (_Float16)v[j];
}

// fc1 bias + jet GELU, f32 (j,n,128) -> f16 same layout.
__global__ void k_head_gelu(const float* __restrict__ H, const float* __restrict__ b1,
                            _Float16* __restrict__ Hh) {
    int idx = blockIdx.x * 256 + threadIdx.x;
    if (idx >= NPTS * 128) return;
    int n = idx / 128, k = idx % 128;
    float v[NJ];
    for (int j = 0; j < NJ; ++j)
        v[j] = H[((size_t)j * NPTS + n) * 128 + k] + ((j == 0) ? b1[k] : 0.f);
    float z   = v[0];
    float Phi = 0.5f * (1.f + erff(z * 0.70710678f));
    float phi = 0.39894228f * __expf(-0.5f * z * z);
    float g1  = Phi + z * phi;
    float g2  = 2.f * phi - z * z * phi;
    float dx = v[1], dy = v[2], dxx = v[3], dyy = v[4];
    v[0] = z * Phi;
    v[1] = g1 * dx;
    v[2] = g1 * dy;
    v[3] = g2 * dx * dx + g1 * dxx;
    v[4] = g2 * dy * dy + g1 * dyy;
    for (int j = 0; j < NJ; ++j) Hh[((size_t)j * NPTS + n) * 128 + k] = (_Float16)v[j];
}

// fc2, field column only (out[...,0]): F[j*N+n].
__global__ void k_fc2(const _Float16* __restrict__ Hh, const float* __restrict__ w2,
                      const float* __restrict__ b2, float* __restrict__ F) {
    int idx = blockIdx.x * 256 + threadIdx.x;
    if (idx >= NJ * NPTS) return;
    int j = idx / NPTS, n = idx % NPTS;
    float s = (j == 0) ? b2[0] : 0.f;
    const _Float16* h = &Hh[((size_t)j * NPTS + n) * 128];
    for (int k = 0; k < 128; ++k) s += (float)h[k] * w2[k * 2];
    F[idx] = s;
}

// ------------------------------ loss ---------------------------------------

__global__ __launch_bounds__(256)
void k_loss_partial(const float* __restrict__ Fu, const float* __restrict__ Fv,
                    const float* __restrict__ Fp, const float* __restrict__ ud,
                    const float* __restrict__ vd, const float* __restrict__ pd,
                    const int* __restrict__ mask, float* __restrict__ part) {
    float a[7] = {0, 0, 0, 0, 0, 0, 0};
    for (int n = blockIdx.x * 256 + threadIdx.x; n < NPTS; n += gridDim.x * 256) {
        float u = Fu[n], ux = Fu[NPTS + n], uy = Fu[2 * NPTS + n],
              uxx = Fu[3 * NPTS + n], uyy = Fu[4 * NPTS + n];
        float v = Fv[n], vx = Fv[NPTS + n], vy = Fv[2 * NPTS + n],
              vxx = Fv[3 * NPTS + n], vyy = Fv[4 * NPTS + n];
        float p = Fp[n], px = Fp[NPTS + n], py = Fp[2 * NPTS + n];
        float fu = v * uy - u * vy + px - NU_C * (uxx + uyy);
        float fv = u * vx - v * ux + py - NU_C * (vxx + vyy);
        float ic = ux + vy;
        float m = (float)mask[n];
        float du = ud[n] - u, dv = vd[n] - v, dp = pd[n] - p;
        a[0] += m; a[1] += m * du * du; a[2] += m * dv * dv; a[3] += m * dp * dp;
        a[4] += fu * fu; a[5] += fv * fv; a[6] += ic * ic;
    }
    __shared__ float sm[256];
    for (int k = 0; k < 7; ++k) {
        sm[threadIdx.x] = a[k]; __syncthreads();
        for (int s = 128; s > 0; s >>= 1) {
            if (threadIdx.x < s) sm[threadIdx.x] += sm[threadIdx.x + s];
            __syncthreads();
        }
        if (threadIdx.x == 0) part[blockIdx.x * 7 + k] = sm[0];
        __syncthreads();
    }
}

__global__ __launch_bounds__(256)
void k_loss_final(const float* __restrict__ part, int nb, float* __restrict__ out) {
    float a[7] = {0, 0, 0, 0, 0, 0, 0};
    for (int i = threadIdx.x; i < nb; i += 256)
        for (int k = 0; k < 7; ++k) a[k] += part[i * 7 + k];
    __shared__ float sm[256];
    float r[7];
    for (int k = 0; k < 7; ++k) {
        sm[threadIdx.x] = a[k]; __syncthreads();
        for (int s = 128; s > 0; s >>= 1) {
            if (threadIdx.x < s) sm[threadIdx.x] += sm[threadIdx.x + s];
            __syncthreads();
        }
        r[k] = sm[0]; __syncthreads();
    }
    if (threadIdx.x == 0) {
        float invN = 1.f / (float)NPTS;
        float ubc = r[1] / r[0], vbc = r[2] / r[0], pbc = r[3] / r[0];
        float ful = r[4] * invN, fvl = r[5] * invN, icl = r[6] * invN;
        float rans = ful + fvl;
        out[0] = ubc + rans + icl; out[1] = vbc + rans + icl; out[2] = pbc + rans + icl;
        out[3] = ubc; out[4] = vbc; out[5] = pbc;
        out[6] = ful; out[7] = fvl; out[8] = icl;
    }
}

// ------------------------------ workspace map ------------------------------

constexpr size_t ALGN = 256;
constexpr size_t al(size_t x) { return (x + ALGN - 1) & ~(ALGN - 1); }

constexpr size_t SZ_ACT  = (size_t)NJ * NPTS * CP * 2;       // 40 MB
constexpr size_t SZ_F    = (size_t)NJ * NPTS * 4;            // field jets, f32
constexpr size_t SZ_PART = 256 * 7 * 4;
constexpr size_t SZ_PW   = 4 * 64 * 64 * 2;
constexpr size_t SZ_FC1  = 64 * 128 * 2;
constexpr size_t SZ_DYF  = 512 * 32 * 2,  SZ_DXF = 128 * 64 * 2;
constexpr size_t SZ_DXI  = 64 * 256 * 2,  SZ_DYI = 32 * 512 * 2;

constexpr size_t OFF_ACT_A = 0;
constexpr size_t OFF_ACT_B = al(OFF_ACT_A + SZ_ACT);
constexpr size_t OFF_FU    = al(OFF_ACT_B + SZ_ACT);
constexpr size_t OFF_FV    = al(OFF_FU + SZ_F);
constexpr size_t OFF_FP    = al(OFF_FV + SZ_F);
constexpr size_t OFF_PART  = al(OFF_FP + SZ_F);
constexpr size_t OFF_PW    = al(OFF_PART + SZ_PART);
constexpr size_t OFF_FC1   = al(OFF_PW + SZ_PW);
constexpr size_t OFF_DYF   = al(OFF_FC1 + SZ_FC1);
constexpr size_t OFF_DXF   = al(OFF_DYF + SZ_DYF);
constexpr size_t OFF_DXI   = al(OFF_DXF + SZ_DXF);
constexpr size_t OFF_DYI   = al(OFF_DXI + SZ_DXI);
constexpr size_t OFF_POOL  = al(OFF_DYI + SZ_DYI);
// pool internals (spectral-conv phase):
constexpr size_t PG1 = 0;
constexpr size_t PY1 = al(PG1 + (size_t)40960 * 512 * 2);
constexpr size_t PG2 = al(PY1 + (size_t)40960 * 32 * 4);
constexpr size_t PY2 = al(PG2 + (size_t)10240 * 128 * 2);
constexpr size_t PG3 = al(PY2 + (size_t)10240 * 64 * 4);
constexpr size_t PY3 = al(PG3 + (size_t)10240 * 64 * 2);
constexpr size_t PG4 = al(PY3 + (size_t)10240 * 256 * 4);
constexpr size_t PX1 = al(PG4 + (size_t)40960 * 32 * 2);
constexpr size_t PX2 = al(PX1 + (size_t)40960 * 512 * 4);
// head phase reuses pool: H f32 at pool+0 (168 MB), Hf16 after it (84 MB).
constexpr size_t PHH  = 0;
constexpr size_t PHH2 = al(PHH + (size_t)NJ * NPTS * 128 * 4);

// ------------------------------ host side ----------------------------------

struct Net {
    const float *fc0_w, *fc0_b;
    const float *w1r[4], *w1i[4], *w2r[4], *w2i[4];
    const float *pw_w[4], *pw_b[4];
    const float *fc1_w, *fc1_b, *fc2_w, *fc2_b;
};

// Param dicts assumed flattened in insertion order (fc0_w, fc0_b,
// then per-layer w1r,w1i,w2r,w2i,pw_w,pw_b, then fc1_w,fc1_b,fc2_w,fc2_b).
static Net unpack(void* const* d_in, int base) {
    Net q; int k = base;
    q.fc0_w = (const float*)d_in[k++]; q.fc0_b = (const float*)d_in[k++];
    for (int l = 0; l < 4; ++l) {
        q.w1r[l] = (const float*)d_in[k++]; q.w1i[l] = (const float*)d_in[k++];
        q.w2r[l] = (const float*)d_in[k++]; q.w2i[l] = (const float*)d_in[k++];
        q.pw_w[l] = (const float*)d_in[k++]; q.pw_b[l] = (const float*)d_in[k++];
    }
    q.fc1_w = (const float*)d_in[k++]; q.fc1_b = (const float*)d_in[k++];
    q.fc2_w = (const float*)d_in[k++]; q.fc2_b = (const float*)d_in[k++];
    return q;
}

static inline void gemm(hipStream_t s, const _Float16* A, const _Float16* B, float* C,
                        int M, int K, int Ncol, int lda, int ldb, int ldc) {
    dim3 g((M + TM - 1) / TM, (Ncol + TN - 1) / TN);
    k_gemm_f16f32<<<g, dim3(256), 0, s>>>(A, B, C, M, K, Ncol, lda, ldb, ldc, 0);
}

static void run_network(hipStream_t s, const Net& P, const float* x, const float* y,
                        char* w, float* Fout) {
    _Float16* ACT_A = (_Float16*)(w + OFF_ACT_A);
    _Float16* ACT_B = (_Float16*)(w + OFF_ACT_B);
    _Float16* PW    = (_Float16*)(w + OFF_PW);
    _Float16* FC1H  = (_Float16*)(w + OFF_FC1);
    _Float16* DYF   = (_Float16*)(w + OFF_DYF);
    _Float16* DXF   = (_Float16*)(w + OFF_DXF);
    _Float16* DXI   = (_Float16*)(w + OFF_DXI);
    _Float16* DYI   = (_Float16*)(w + OFF_DYI);
    char* pool = w + OFF_POOL;
    _Float16* G1 = (_Float16*)(pool + PG1);
    float*    Y1 = (float*)   (pool + PY1);
    _Float16* G2 = (_Float16*)(pool + PG2);
    float*    Y2 = (float*)   (pool + PY2);
    _Float16* G3 = (_Float16*)(pool + PG3);
    float*    Y3 = (float*)   (pool + PY3);
    _Float16* G4 = (_Float16*)(pool + PG4);
    float*    X1 = (float*)   (pool + PX1);
    float*    X2 = (float*)   (pool + PX2);
    float*    HH = (float*)   (pool + PHH);
    _Float16* HH2= (_Float16*)(pool + PHH2);

    for (int l = 0; l < 4; ++l)
        k_pad_f16<<<(64 * 64 + 255) / 256, 256, 0, s>>>(P.pw_w[l], PW + l * 64 * 64,
                                                        WID, WID, 64, 64);
    k_pad_f16<<<(64 * 128 + 255) / 256, 256, 0, s>>>(P.fc1_w, FC1H, WID, 128, 64, 128);

    k_lift<<<(NPTS + 255) / 256, 256, 0, s>>>(x, y, P.fc0_w, P.fc0_b, ACT_A);

    _Float16* cur = ACT_A;
    _Float16* nxt = ACT_B;
    const int NE1 = NJ * CP * kS1 * kS2;     // 20,971,520
    const int NE2 = NJ * CP * 32 * kS1;      //  1,310,720
    for (int l = 0; l < 4; ++l) {
        // spectral conv as truncated separable DFT GEMMs
        k_perm_fwd<<<NE1 / 256, 256, 0, s>>>(cur, G1);
        gemm(s, G1, DYF, Y1, 40960, 512, 32, 512, 32, 32);
        k_perm_y1g2<<<NE2 / 256, 256, 0, s>>>(Y1, G2);
        gemm(s, G2, DXF, Y2, 10240, 128, 64, 128, 64, 64);
        k_fill_f16<<<(10240 * 64 + 255) / 256, 256, 0, s>>>(G3, 10240 * 64);
        k_modemix<<<NJ * 24 * M2C, 64, 0, s>>>(Y2, P.w1r[l], P.w1i[l],
                                               P.w2r[l], P.w2i[l], G3);
        gemm(s, G3, DXI, Y3, 10240, 64, 256, 64, 256, 256);
        k_perm_y3g4<<<NE2 / 256, 256, 0, s>>>(Y3, G4);
        gemm(s, G4, DYI, X1, 40960, 32, 512, 32, 512, 512);
        // pointwise conv over the jet-folded rows
        gemm(s, cur, PW + l * 64 * 64, X2, NJ * NPTS, 64, 64, 64, 64, 64);
        k_combine_gelu<<<(NPTS * CP + 255) / 256, 256, 0, s>>>(X1, X2, P.pw_b[l],
                                                               nxt, l < 3 ? 1 : 0);
        _Float16* tp = cur; cur = nxt; nxt = tp;
    }
    // head: fc1 (WMMA), jet-GELU, fc2 field column
    gemm(s, cur, FC1H, HH, NJ * NPTS, 64, 128, 64, 128, 128);
    k_head_gelu<<<(NPTS * 128 + 255) / 256, 256, 0, s>>>(HH, P.fc1_b, HH2);
    k_fc2<<<(NJ * NPTS + 255) / 256, 256, 0, s>>>(HH2, P.fc2_w, P.fc2_b, Fout);
}

extern "C" void kernel_launch(void* const* d_in, const int* in_sizes, int n_in,
                              void* d_out, int out_size, void* d_ws, size_t ws_size,
                              hipStream_t stream) {
    (void)in_sizes; (void)n_in; (void)out_size; (void)ws_size;
    char* w = (char*)d_ws;
    const float* x  = (const float*)d_in[0];
    const float* y  = (const float*)d_in[1];
    const float* ud = (const float*)d_in[2];
    const float* vd = (const float*)d_in[3];
    const float* pd = (const float*)d_in[4];
    const int*   bm = (const int*)d_in[5];

    _Float16* DYF = (_Float16*)(w + OFF_DYF);
    _Float16* DXF = (_Float16*)(w + OFF_DXF);
    _Float16* DXI = (_Float16*)(w + OFF_DXI);
    _Float16* DYI = (_Float16*)(w + OFF_DYI);
    k_twiddles<<<64, 256, 0, stream>>>(DYF, DXF, DXI, DYI);

    float* FU = (float*)(w + OFF_FU);
    float* FV = (float*)(w + OFF_FV);
    float* FP = (float*)(w + OFF_FP);
    float* F[3] = {FU, FV, FP};
    for (int net = 0; net < 3; ++net) {
        Net P = unpack(d_in, 6 + net * 30);
        run_network(stream, P, x, y, w, F[net]);
    }

    float* PART = (float*)(w + OFF_PART);
    k_loss_partial<<<256, 256, 0, stream>>>(FU, FV, FP, ud, vd, pd, bm, PART);
    k_loss_final<<<1, 256, 0, stream>>>(PART, 256, (float*)d_out);
}
